// MultiHeadedAttention_57380763074828
// MI455X (gfx1250) — matrix operand order repaired and checked by reading, not compiled, not run on hardware
//
#include <hip/hip_runtime.h>

// ---- CDNA5 WMMA / TDM vector types ----
typedef __attribute__((ext_vector_type(16))) __bf16 v16bf;
typedef __attribute__((ext_vector_type(8)))  __bf16 v8bf;
typedef __attribute__((ext_vector_type(8)))  float  v8f;
typedef __attribute__((ext_vector_type(4)))  unsigned int u32x4;
typedef __attribute__((ext_vector_type(8)))  int i32x8;
typedef __attribute__((ext_vector_type(4)))  int i32x4;

#define MIN_SCORE (-1.0e9f)

#if defined(__gfx1250__) && __has_builtin(__builtin_amdgcn_tensor_load_to_lds)
#define HAVE_TDM 1
#else
#define HAVE_TDM 0
#endif

#if HAVE_TDM
// 2D tile load via Tensor Data Mover: bf16 elements, row-major tensor
// (tensor_w elements per row, pitch elements between rows), tile
// (tile_w x tile_h) starting at gptr, written contiguously to LDS at lds_off.
// D# bitfields per CDNA5 ISA ch.8 (group0: count/lds/global/type; group1:
// data_size=2B, dims, tile dims, dim0 stride). Groups 2/3 unused (2D tile).
__device__ __forceinline__ void tdm_load_tile_bf16(const void* gptr, unsigned lds_off,
                                                   unsigned tensor_w, unsigned tensor_h,
                                                   unsigned tile_w, unsigned tile_h,
                                                   unsigned pitch) {
    unsigned long long ga = (unsigned long long)(uintptr_t)gptr;
    u32x4 g0;
    g0[0] = 1u;                                            // count=1, user descriptor
    g0[1] = lds_off;                                       // LDS byte address
    g0[2] = (unsigned)(ga & 0xffffffffu);                  // global_addr[31:0]
    g0[3] = (unsigned)((ga >> 32) & 0x01ffffffu) | (2u << 30); // addr[56:32] | type=2
    i32x8 g1;
    g1[0] = (int)(1u << 16);                               // data_size=1 (2 bytes)
    g1[1] = (int)((tensor_w & 0xffffu) << 16);             // tensor_dim0[15:0]
    g1[2] = (int)((tensor_w >> 16) | ((tensor_h & 0xffffu) << 16));
    g1[3] = (int)((tensor_h >> 16) | (tile_w << 16));      // tensor_dim1[31:16]|tile_dim0
    g1[4] = (int)(tile_h & 0xffffu);                       // tile_dim1 (tile_dim2=0)
    g1[5] = (int)pitch;                                    // tensor_dim0_stride[31:0]
    g1[6] = 0;
    g1[7] = 0;
    i32x4 z4 = {0, 0, 0, 0};
    i32x8 z8 = {0, 0, 0, 0, 0, 0, 0, 0};
    // 6-arg form (clang-23 / therock-10.0): (g0, g1, g2, g3, g4, cpol)
    __builtin_amdgcn_tensor_load_to_lds(g0, g1, z4, z4, z8, 0);
}
#endif

// =====================================================================
// fp32 -> bf16 conversion (one-time weight pre-pass)
// =====================================================================
__global__ __launch_bounds__(256) void cvt_bf16_kernel(const float* __restrict__ src,
                                                       __bf16* __restrict__ dst, int n) {
    int i = (blockIdx.x * 256 + threadIdx.x) * 4;
    if (i + 3 < n) {
        float4 f = *(const float4*)(src + i);
        dst[i + 0] = (__bf16)f.x;
        dst[i + 1] = (__bf16)f.y;
        dst[i + 2] = (__bf16)f.z;
        dst[i + 3] = (__bf16)f.w;
    }
}

// =====================================================================
// GEMM: out = A @ W^T + bias ; W pre-converted bf16 row-major N x K.
// Double-buffered LDS: TDM (or manual cvt) for tile k+1 in flight while
// tile k is consumed by WMMA; one barrier per k-step.
// ABF16: A is bf16 (TDM-staged) else fp32 (float4 + cvt stage-in).
// OUTMODE 0: fp32 row-major [M,N]; OUTMODE 1: head-split bf16 (S=4096 fixed).
// Block 256 threads (8 waves); C-tile 128x64; wave w -> rows [16w,16w+16).
// =====================================================================
template <bool ABF16, int OUTMODE>
__global__ __launch_bounds__(256) void gemm_kernel(const void* __restrict__ A_,
                                                   const __bf16* __restrict__ W,
                                                   const float* __restrict__ bias,
                                                   void* __restrict__ out_,
                                                   int M, int N, int K) {
    __shared__ __bf16 As[2][128 * 32];   // [buf][row][k], 64B rows
    __shared__ __bf16 Bs[2][64 * 32];    // [buf][n][k]

    const int t       = threadIdx.x;
    const int rowbase = blockIdx.x * 128;
    const int nbase   = blockIdx.y * 64;
    const int wv      = t >> 5;
    const int l       = t & 31;
    const int lr      = l & 15;
    const int hsel    = l >> 4;

    const v8f zero = {0.f, 0.f, 0.f, 0.f, 0.f, 0.f, 0.f, 0.f};
    v8f acc[4] = {zero, zero, zero, zero};

    const float*  Af = (const float*)A_;
    const __bf16* Ab = (const __bf16*)A_;

    // per-thread manual-stage coordinates (fp32 A path)
    const int r  = t >> 1;
    const int kh = (t & 1) * 16;
    // manual W fallback coords
    const int wn = t >> 2;
    const int wq = (t & 3) * 8;

    auto stage_tiles = [&](int k0, int buf) {
#if HAVE_TDM
        if (ABF16) {
            if (wv == 0)
                tdm_load_tile_bf16(Ab + (size_t)rowbase * K + k0,
                                   (unsigned)(uintptr_t)&As[buf][0],
                                   32, 128, 32, 128, (unsigned)K);
        } else {
            const float4* src = (const float4*)(Af + (size_t)(rowbase + r) * K + k0 + kh);
#pragma unroll
            for (int c = 0; c < 4; ++c) {
                float4 f = src[c];
                As[buf][r * 32 + kh + c * 4 + 0] = (__bf16)f.x;
                As[buf][r * 32 + kh + c * 4 + 1] = (__bf16)f.y;
                As[buf][r * 32 + kh + c * 4 + 2] = (__bf16)f.z;
                As[buf][r * 32 + kh + c * 4 + 3] = (__bf16)f.w;
            }
        }
        if (wv == 0)
            tdm_load_tile_bf16(W + (size_t)nbase * K + k0,
                               (unsigned)(uintptr_t)&Bs[buf][0],
                               32, 64, 32, 64, (unsigned)K);
#else
        if (ABF16) {
            const size_t base = (size_t)(rowbase + r) * K + k0 + kh;
#pragma unroll
            for (int e = 0; e < 16; ++e) As[buf][r * 32 + kh + e] = Ab[base + e];
        } else {
            const float4* src = (const float4*)(Af + (size_t)(rowbase + r) * K + k0 + kh);
#pragma unroll
            for (int c = 0; c < 4; ++c) {
                float4 f = src[c];
                As[buf][r * 32 + kh + c * 4 + 0] = (__bf16)f.x;
                As[buf][r * 32 + kh + c * 4 + 1] = (__bf16)f.y;
                As[buf][r * 32 + kh + c * 4 + 2] = (__bf16)f.z;
                As[buf][r * 32 + kh + c * 4 + 3] = (__bf16)f.w;
            }
        }
        {
            const size_t base = (size_t)(nbase + wn) * K + k0 + wq;
#pragma unroll
            for (int e = 0; e < 8; ++e) Bs[buf][wn * 32 + wq + e] = W[base + e];
        }
#endif
    };

    // ---- prologue: tile 0 into buffer 0 ----
    stage_tiles(0, 0);
#if HAVE_TDM
    __builtin_amdgcn_s_wait_tensorcnt(0);
#endif
    __syncthreads();

    const int nk = K >> 5;
    for (int ki = 0; ki < nk; ++ki) {
        const int cur = ki & 1;
        if (ki + 1 < nk) stage_tiles((ki + 1) << 5, (ki + 1) & 1);

        // A fragment (row=lr; contiguous k order: half h holds k 16h..16h+15 as two 16B chunks)
        v16bf a;
        {
            const __bf16* ap = &As[cur][(wv * 16 + lr) * 32];
            const int kb = hsel ? 16 : 0;
            v8bf lo = *(const v8bf*)(ap + kb);
            v8bf hi = *(const v8bf*)(ap + kb + 8);
#pragma unroll
            for (int j = 0; j < 8; ++j) { a[j] = lo[j]; a[8 + j] = hi[j]; }
        }
#pragma unroll
        for (int ct = 0; ct < 4; ++ct) {
            const __bf16* bp = &Bs[cur][(ct * 16 + lr) * 32 + (hsel ? 16 : 0)];
            v16bf bfrag = *(const v16bf*)bp;
            acc[ct] = __builtin_amdgcn_wmma_f32_16x16x32_bf16(
                false, a, false, bfrag, (short)0, acc[ct], false, false);
        }
#if HAVE_TDM
        __builtin_amdgcn_s_wait_tensorcnt(0);   // next-tile DMA arrived (wave 0)
#endif
        __syncthreads();                        // all waves done with buf[cur]
    }

#pragma unroll
    for (int ct = 0; ct < 4; ++ct) {
        const int gn = nbase + ct * 16 + lr;
        const float bv = bias[gn];
#pragma unroll
        for (int vv = 0; vv < 8; ++vv) {
            const int gm  = rowbase + wv * 16 + vv + 8 * hsel;
            const float val = acc[ct][vv] + bv;
            if (OUTMODE == 0) {
                ((float*)out_)[(size_t)gm * N + gn] = val;
            } else {
                const int bb = gm >> 12;          // S = 4096
                const int s  = gm & 4095;
                const int h  = gn >> 6;
                const int hd = gn & 63;
                ((__bf16*)out_)[(((size_t)(bb * 8 + h) << 12) + s) * 64 + hd] = (__bf16)val;
            }
        }
    }
}

// =====================================================================
// Sliding-window attention: one wave per (b,h, 16-query tile).
// Window keys [qbase-16, qbase+32): 3 key tiles, offsets [-16,16].
// =====================================================================
__global__ __launch_bounds__(32) void attn_kernel(const __bf16* __restrict__ q,
                                                  const __bf16* __restrict__ k,
                                                  const __bf16* __restrict__ v,
                                                  const unsigned char* __restrict__ mask,
                                                  __bf16* __restrict__ out, int S) {
    __shared__ __bf16 Qs[16 * 64];   // [qrow][hd]
    __shared__ __bf16 Ks[48 * 64];   // [key][hd]
    __shared__ __bf16 Vt[64 * 64];   // [hd][key] (keys 48..63 zero)
    __shared__ __bf16 Ps[16 * 64];   // probs [qrow][key_local] (48..63 zero)

    const int qt = blockIdx.x;
    const int bh = blockIdx.y;
    const int b  = bh >> 3;
    const int h  = bh & 7;
    const int l  = threadIdx.x;
    const int lr = l & 15;
    const int hsel = l >> 4;
    const int qbase = qt * 16;
    const int kb0 = qbase - 16;

    const __bf16* qhead = q + (size_t)bh * S * 64;
    const __bf16* khead = k + (size_t)bh * S * 64;
    const __bf16* vhead = v + (size_t)bh * S * 64;

    const bool interior = (kb0 >= 0) && (kb0 + 48 <= S);

#if HAVE_TDM
    // Q tile (16x64, always in range) via TDM
    tdm_load_tile_bf16(qhead + (size_t)qbase * 64, (unsigned)(uintptr_t)Qs,
                       64, (unsigned)S, 64, 16, 64);
    if (interior)   // K window 48x64 via TDM when fully in range
        tdm_load_tile_bf16(khead + (size_t)kb0 * 64, (unsigned)(uintptr_t)Ks,
                           64, (unsigned)S, 64, 48, 64);
#else
#pragma unroll
    for (int i = 0; i < 32; ++i) {
        const int idx = l + i * 32;
        Qs[idx] = qhead[(size_t)qbase * 64 + idx];
    }
#endif
    if (!interior) {
#pragma unroll
        for (int i = 0; i < 96; ++i) {
            const int idx = l + i * 32;
            const int row = idx >> 6, c = idx & 63;
            const int kr = kb0 + row;
            Ks[idx] = (kr >= 0 && kr < S) ? khead[(size_t)kr * 64 + c] : (__bf16)0.f;
        }
    }
#if !HAVE_TDM
    else {
#pragma unroll
        for (int i = 0; i < 96; ++i) {
            const int idx = l + i * 32;
            Ks[idx] = khead[(size_t)kb0 * 64 + idx];
        }
    }
#endif
    // V window transposed into [hd][key] (overlaps with in-flight TDM)
#pragma unroll
    for (int i = 0; i < 96; ++i) {
        const int idx = l + i * 32;
        const int row = idx >> 6, c = idx & 63;
        const int kr = kb0 + row;
        Vt[c * 64 + row] = (kr >= 0 && kr < S) ? vhead[(size_t)kr * 64 + c] : (__bf16)0.f;
    }
#pragma unroll
    for (int i = 0; i < 32; ++i) {
        const int idx = l + i * 32;
        Vt[(idx >> 4) * 64 + 48 + (idx & 15)] = (__bf16)0.f;
    }
#if HAVE_TDM
    __builtin_amdgcn_s_wait_tensorcnt(0);
#endif
    __syncthreads();

    const v8f zero = {0.f, 0.f, 0.f, 0.f, 0.f, 0.f, 0.f, 0.f};
    v8f st[3] = {zero, zero, zero};

#pragma unroll
    for (int ko = 0; ko < 64; ko += 32) {
        v16bf a;
        {
            const __bf16* ap = &Qs[lr * 64 + ko];
            const int kb = hsel ? 16 : 0;
            v8bf lo = *(const v8bf*)(ap + kb);
            v8bf hi = *(const v8bf*)(ap + kb + 8);
#pragma unroll
            for (int j = 0; j < 8; ++j) { a[j] = lo[j]; a[8 + j] = hi[j]; }
        }
#pragma unroll
        for (int t = 0; t < 3; ++t) {
            const __bf16* bp = &Ks[(t * 16 + lr) * 64 + ko + (hsel ? 16 : 0)];
            v16bf bfrag = *(const v16bf*)bp;
            st[t] = __builtin_amdgcn_wmma_f32_16x16x32_bf16(
                false, a, false, bfrag, (short)0, st[t], false, false);
        }
    }

    // mask + softmax (rows in VGPRs, keys across 16 lanes x 3 tiles)
    const unsigned char* mrow = mask + (size_t)b * S;
    float sc[3][8];
#pragma unroll
    for (int t = 0; t < 3; ++t) {
        const int ki = kb0 + t * 16 + lr;
        const bool kvalid = (ki >= 0 && ki < S);
        int kc = ki < 0 ? 0 : (ki >= S ? S - 1 : ki);   // clamped, branch-free load
        const bool masked = (mrow[kc] != 0) || !kvalid;
#pragma unroll
        for (int vv = 0; vv < 8; ++vv) {
            const int m   = vv + 8 * hsel;
            const int off = ki - (qbase + m);
            const bool ok = !masked && off >= -16 && off <= 16;
            sc[t][vv] = ok ? st[t][vv] : MIN_SCORE;
        }
    }
#pragma unroll
    for (int vv = 0; vv < 8; ++vv) {
        float mx = fmaxf(fmaxf(sc[0][vv], sc[1][vv]), sc[2][vv]);
#pragma unroll
        for (int d = 1; d < 16; d <<= 1) mx = fmaxf(mx, __shfl_xor(mx, d, 32));
        float s = 0.f;
#pragma unroll
        for (int t = 0; t < 3; ++t) { sc[t][vv] = __expf(sc[t][vv] - mx); s += sc[t][vv]; }
#pragma unroll
        for (int d = 1; d < 16; d <<= 1) s += __shfl_xor(s, d, 32);
        const float inv = 1.f / s;
#pragma unroll
        for (int t = 0; t < 3; ++t) sc[t][vv] *= inv;
    }

#pragma unroll
    for (int t = 0; t < 3; ++t)
#pragma unroll
        for (int vv = 0; vv < 8; ++vv)
            Ps[(vv + 8 * hsel) * 64 + t * 16 + lr] = (__bf16)sc[t][vv];
#pragma unroll
    for (int i = 0; i < 8; ++i) {
        const int idx = l + i * 32;
        Ps[(idx >> 4) * 64 + 48 + (idx & 15)] = (__bf16)0.f;
    }
    __syncthreads();

    const size_t orow_base = (size_t)b * S + qbase;
#pragma unroll
    for (int ct = 0; ct < 4; ++ct) {
        v8f o = zero;
#pragma unroll
        for (int ko = 0; ko < 64; ko += 32) {
            v16bf a;
            {
                const __bf16* ap = &Ps[lr * 64 + ko];
                const int kb = hsel ? 16 : 0;
                v8bf lo = *(const v8bf*)(ap + kb);
                v8bf hi = *(const v8bf*)(ap + kb + 8);
#pragma unroll
                for (int j = 0; j < 8; ++j) { a[j] = lo[j]; a[8 + j] = hi[j]; }
            }
            const __bf16* bp = &Vt[(ct * 16 + lr) * 64 + ko + (hsel ? 16 : 0)];
            v16bf bfrag = *(const v16bf*)bp;
            o = __builtin_amdgcn_wmma_f32_16x16x32_bf16(
                false, a, false, bfrag, (short)0, o, false, false);
        }
#pragma unroll
        for (int vv = 0; vv < 8; ++vv) {
            const int m = vv + 8 * hsel;
            out[(orow_base + m) * 512 + h * 64 + ct * 16 + lr] = (__bf16)o[vv];
        }
    }
}

// =====================================================================
extern "C" void kernel_launch(void* const* d_in, const int* in_sizes, int n_in,
                              void* d_out, int out_size, void* d_ws, size_t ws_size,
                              hipStream_t stream) {
    const float* query = (const float*)d_in[0];
    const float* key   = (const float*)d_in[1];
    const float* value = (const float*)d_in[2];
    const unsigned char* mask = (const unsigned char*)d_in[3];
    const float* Wq = (const float*)d_in[4];
    const float* bq = (const float*)d_in[5];
    const float* Wk = (const float*)d_in[6];
    const float* bk = (const float*)d_in[7];
    const float* Wv = (const float*)d_in[8];
    const float* bv = (const float*)d_in[9];
    const float* Wo = (const float*)d_in[10];
    const float* bo = (const float*)d_in[11];

    const int B = 8, S = 4096, D = 512;
    const int M = B * S;
    const int WN = D * D;                      // 262144 per weight matrix

    // workspace: 4 bf16 activation buffers (M*D each) + 4 bf16 weight copies
    __bf16* qws = (__bf16*)d_ws;
    const size_t seg = (size_t)M * D;
    __bf16* kws = qws + seg;
    __bf16* vws = kws + seg;
    __bf16* aws = vws + seg;
    __bf16* wqb = aws + seg;
    __bf16* wkb = wqb + WN;
    __bf16* wvb = wkb + WN;
    __bf16* wob = wvb + WN;

    // one-time weight conversion fp32 -> bf16
    dim3 cg(WN / (256 * 4)), cb(256);
    cvt_bf16_kernel<<<cg, cb, 0, stream>>>(Wq, wqb, WN);
    cvt_bf16_kernel<<<cg, cb, 0, stream>>>(Wk, wkb, WN);
    cvt_bf16_kernel<<<cg, cb, 0, stream>>>(Wv, wvb, WN);
    cvt_bf16_kernel<<<cg, cb, 0, stream>>>(Wo, wob, WN);

    dim3 gg(M / 128, D / 64), gb(256);
    gemm_kernel<false, 1><<<gg, gb, 0, stream>>>(query, wqb, bq, qws, M, D, D);
    gemm_kernel<false, 1><<<gg, gb, 0, stream>>>(key,   wkb, bk, kws, M, D, D);
    gemm_kernel<false, 1><<<gg, gb, 0, stream>>>(value, wvb, bv, vws, M, D, D);

    dim3 ga(S / 16, B * 8);
    attn_kernel<<<ga, dim3(32), 0, stream>>>(qws, kws, vws, mask, aws, S);

    gemm_kernel<true, 0><<<gg, gb, 0, stream>>>(aws, wob, bo, d_out, M, D, D);
}
